// MultiHeadAttention_74921409511792
// MI455X (gfx1250) — matrix operand order repaired and checked by reading, not compiled, not run on hardware
//
#include <hip/hip_runtime.h>
#include <hip/hip_bf16.h>
#include <stdint.h>

// CDNA5 / gfx1250 wave32 WMMA types
typedef __attribute__((ext_vector_type(16))) __bf16 v16bf;
typedef __attribute__((ext_vector_type(8)))  float  v8f;

union Frag {
  v16bf v;
  unsigned int u[8];
};

__device__ __forceinline__ unsigned short f2bf(float x) {
  unsigned int u = __float_as_uint(x);
  u += 0x7FFFu + ((u >> 16) & 1u);   // round-to-nearest-even
  return (unsigned short)(u >> 16);
}

// 32-bit LDS byte offset of a __shared__ object (generic -> addrspace(3))
typedef __attribute__((address_space(3))) const void lds_cvoid;
__device__ __forceinline__ unsigned lds_addr32(const void* p) {
  return (unsigned)(unsigned long long)(lds_cvoid*)p;
}

// Async global->LDS copy, 16 bytes per lane. Tracked by ASYNCcnt.
// GVS addressing: 64-bit SGPR base + per-lane 32-bit byte offset (ISA 08 §4.4).
__device__ __forceinline__ void async_load_b128(unsigned lds_off, unsigned gvoff,
                                                const void* sbase) {
  asm volatile("global_load_async_to_lds_b128 %0, %1, %2 offset:0"
               :
               : "v"(lds_off), "v"(gvoff), "s"(sbase)
               : "memory");
}
__device__ __forceinline__ void wait_async0() {
  asm volatile("s_wait_asynccnt 0" ::: "memory");
}

// A-matrix (16x32 bf16) fragment: lane = row (l&15); K pairs at
// koff + {0,2,4,6,16,18,20,22}, koff = 8 for lanes 16-31. (ISA 7.12.2)
__device__ __forceinline__ v16bf a_frag_lds(const unsigned short* rowPtr, int koff) {
  Frag f;
#pragma unroll
  for (int p = 0; p < 8; ++p) {
    int k = koff + ((p < 4) ? (2 * p) : (2 * p + 8));
    f.u[p] = *(const unsigned int*)(rowPtr + k);
  }
  return f.v;
}

// B-matrix (32x16 bf16) fragment: lane = column (l&15); K pairs at
// kbase/2 + {0,2,4,6,16,18,20,22}, kbase/2 = 8 for lanes 16-31 (same as A).
__device__ __forceinline__ v16bf b_frag_lds(const unsigned short* colPtr, int kbase) {
  Frag f;
#pragma unroll
  for (int p = 0; p < 8; ++p)
    f.u[p] = *(const unsigned int*)(colPtr + (kbase >> 1) + ((p < 4) ? (2 * p) : (2 * p + 8)));
  return f.v;
}

// ---------------------------------------------------------------------------
// GEMM: out[M,Nout] = A[M,K] * W[Nout,K]^T + bias
// Block tile 128x64, BK=32, 8 waves each computing a 32x32 (2x2 WMMA) tile.
// EPI==0: A fp32 (x), scatter epilogue -> q/k/v bf16 [B,H,N,D]
// EPI==1: A bf16 (attn out, async-staged), epilogue -> fp32 row-major out
// ---------------------------------------------------------------------------
template <int EPI>
__global__ __launch_bounds__(256) void gemm_wmma_bf16(
    const float* __restrict__ Afp, const unsigned short* __restrict__ Abf,
    const float* __restrict__ W, const float* __restrict__ bias,
    int M, int Nout, int Kdim,
    unsigned short* __restrict__ qo, unsigned short* __restrict__ ko,
    unsigned short* __restrict__ vo, float* __restrict__ outf) {
  __shared__ unsigned short As[128][40] __attribute__((aligned(16)));
  __shared__ unsigned short Bs[64][40] __attribute__((aligned(16)));

  const int t = threadIdx.x;
  const int lane = t & 31;
  const int wv = t >> 5;
  const int l15 = lane & 15;
  const int hi = lane >> 4;
  const int wm = (wv & 3) * 32;     // wave row offset in block tile
  const int wn = (wv >> 2) * 32;    // wave col offset in block tile
  const int koffA = hi ? 8 : 0;
  const int kbaseB = hi ? 16 : 0;
  const int rowBase = blockIdx.y * 128;
  const int colBase = blockIdx.x * 64;

  v8f acc[2][2] = {};

  for (int k0 = 0; k0 < Kdim; k0 += 32) {
    // stage A tile (128x32) into LDS as bf16
    {
      int r = t >> 1;
      int c = (t & 1) * 16;
      if (EPI == 0) {
        const float* src = Afp + (size_t)(rowBase + r) * Kdim + k0 + c;
#pragma unroll
        for (int i = 0; i < 16; i += 4) {
          float4 f = *(const float4*)(src + i);
          As[r][c + i + 0] = f2bf(f.x);
          As[r][c + i + 1] = f2bf(f.y);
          As[r][c + i + 2] = f2bf(f.z);
          As[r][c + i + 3] = f2bf(f.w);
        }
      } else {
        // async global->LDS, 2x16B per thread, no VGPR round-trip
        unsigned loff = lds_addr32(&As[r][c]);
        unsigned goff = (unsigned)(((size_t)(rowBase + r) * Kdim + k0 + c) * 2);
        async_load_b128(loff, goff, Abf);
        async_load_b128(loff + 16u, goff + 16u, Abf);
      }
    }
    // stage B tile: rows of W (64x32) into LDS as bf16, layout [n][k]
    {
      int r = t >> 2;
      int c = (t & 3) * 8;
      const float* src = W + (size_t)(colBase + r) * Kdim + k0 + c;
      float4 f0 = *(const float4*)(src);
      float4 f1 = *(const float4*)(src + 4);
      Bs[r][c + 0] = f2bf(f0.x); Bs[r][c + 1] = f2bf(f0.y);
      Bs[r][c + 2] = f2bf(f0.z); Bs[r][c + 3] = f2bf(f0.w);
      Bs[r][c + 4] = f2bf(f1.x); Bs[r][c + 5] = f2bf(f1.y);
      Bs[r][c + 6] = f2bf(f1.z); Bs[r][c + 7] = f2bf(f1.w);
    }
    if (EPI != 0) wait_async0();
    __syncthreads();

    Frag a0, a1, b0, b1;
    a0.v = a_frag_lds(&As[wm + l15][0], koffA);
    a1.v = a_frag_lds(&As[wm + 16 + l15][0], koffA);
    b0.v = b_frag_lds(&Bs[wn + l15][0], kbaseB);
    b1.v = b_frag_lds(&Bs[wn + 16 + l15][0], kbaseB);

    acc[0][0] = __builtin_amdgcn_wmma_f32_16x16x32_bf16(false, a0.v, false, b0.v, (short)0, acc[0][0], false, false);
    acc[0][1] = __builtin_amdgcn_wmma_f32_16x16x32_bf16(false, a0.v, false, b1.v, (short)0, acc[0][1], false, false);
    acc[1][0] = __builtin_amdgcn_wmma_f32_16x16x32_bf16(false, a1.v, false, b0.v, (short)0, acc[1][0], false, false);
    acc[1][1] = __builtin_amdgcn_wmma_f32_16x16x32_bf16(false, a1.v, false, b1.v, (short)0, acc[1][1], false, false);
    __syncthreads();
  }

  // epilogue: C/D layout -> element r of v8f is row (r + 8*hi), col (l&15)
#pragma unroll
  for (int i = 0; i < 2; ++i) {
#pragma unroll
    for (int j = 0; j < 2; ++j) {
      int colB = colBase + wn + j * 16 + l15;
      float bv = bias[colB];
#pragma unroll
      for (int r = 0; r < 8; ++r) {
        int rowB = rowBase + wm + i * 16 + r + hi * 8;
        float val = acc[i][j][r] + bv;
        if (EPI == 0) {
          // qkv column -> (sel, head, d); scatter to [B,H,N,D] bf16
          int bb = rowB >> 10, nn = rowB & 1023;
          int ssel = colB / 768;
          int rem = colB - ssel * 768;
          int hh = rem >> 6, dd = rem & 63;
          unsigned short* dst = (ssel == 0) ? qo : (ssel == 1) ? ko : vo;
          dst[((((size_t)bb * 12 + hh) << 10) + nn) * 64 + dd] = f2bf(val);
        } else {
          outf[(size_t)rowB * Nout + colB] = val;
        }
      }
    }
  }
}

// ---------------------------------------------------------------------------
// Flash attention with ALiBi. Q/K/V bf16 [B,H,N,64]. One block = 128 query
// rows of one (b,h); each wave owns 16 rows. Online softmax over 16 key
// tiles of 64. K tile staged with async global->LDS copies (ASYNCcnt path);
// V tile transposed through VGPRs for the B-operand layout.
// Output -> bf16 [B,N,C] (transposed back for the projection).
// ---------------------------------------------------------------------------
__global__ __launch_bounds__(256) void flash_attn_alibi(
    const unsigned short* __restrict__ Qp, const unsigned short* __restrict__ Kp,
    const unsigned short* __restrict__ Vp, unsigned short* __restrict__ AOp) {
  __shared__ unsigned short Ks[64][72]    __attribute__((aligned(16)));  // [key][d]
  __shared__ unsigned short Vs[64][72]    __attribute__((aligned(16)));  // [d][key]
  __shared__ unsigned short Ps[8][16][64] __attribute__((aligned(16)));  // per-wave P

  const int t = threadIdx.x;
  const int lane = t & 31;
  const int wv = t >> 5;
  const int l15 = lane & 15;
  const int hi = lane >> 4;
  const int koff = hi ? 8 : 0;
  const int kb16 = hi ? 16 : 0;

  const int bh = blockIdx.y;
  const int bidx = bh / 12;
  const int hidx = bh - bidx * 12;
  const int q0 = blockIdx.x * 128;
  const float slope = exp2f(-(8.0f / 12.0f) * (float)(hidx + 1));

  const size_t bhOff = ((size_t)bh << 10) * 64;
  const unsigned short* Qb = Qp + bhOff;
  const unsigned short* Kb = Kp + bhOff;
  const unsigned short* Vb = Vp + bhOff;

  // Q A-fragments for this wave's 16 rows (D=64 -> two 16x32 frags), kept in VGPRs
  Frag qf[2];
  {
    const unsigned short* qrp = Qb + (size_t)(q0 + wv * 16 + l15) * 64;
#pragma unroll
    for (int c = 0; c < 2; ++c)
#pragma unroll
      for (int p = 0; p < 8; ++p) {
        int k = 32 * c + koff + ((p < 4) ? (2 * p) : (2 * p + 8));
        qf[c].u[p] = *(const unsigned int*)(qrp + k);
      }
  }

  v8f o[4] = {};
  float mrow[8], lrow[8];
#pragma unroll
  for (int r = 0; r < 8; ++r) { mrow[r] = -1e30f; lrow[r] = 0.0f; }

  for (int kt = 0; kt < 16; ++kt) {
    const int key0 = kt * 64;
    {
      int r = t >> 2;
      int c = (t & 3) * 16;
      // K tile: async global->LDS, 2x16B per thread
      unsigned loff = lds_addr32(&Ks[r][c]);
      unsigned goff = (unsigned)(((size_t)(key0 + r) * 64 + c) * 2);
      async_load_b128(loff, goff, Kb);
      async_load_b128(loff + 16u, goff + 16u, Kb);
      // V tile: load through VGPRs and transpose for the B-operand
      const unsigned short* vsrc = Vb + (size_t)(key0 + r) * 64 + c;
      union { uint4 q4[2]; unsigned short s16[16]; } vt;
      vt.q4[0] = *(const uint4*)vsrc;
      vt.q4[1] = *(const uint4*)(vsrc + 8);
#pragma unroll
      for (int i = 0; i < 16; ++i) Vs[c + i][r] = vt.s16[i];
    }
    wait_async0();
    __syncthreads();

    // S = Q * K^T  (contraction over d; B[k=d][n=key] read from Ks[key][d])
    v8f sc[4];
#pragma unroll
    for (int j = 0; j < 4; ++j) {
      v8f z = {};
#pragma unroll
      for (int c = 0; c < 2; ++c) {
        v16bf bf = b_frag_lds(&Ks[j * 16 + l15][32 * c], kb16);
        z = __builtin_amdgcn_wmma_f32_16x16x32_bf16(false, qf[c].v, false, bf, (short)0, z, false, false);
      }
      sc[j] = z;
    }

    // scale + ALiBi + online softmax (row stats reduce across the 16-lane half)
    const float rowf = (float)(q0 + wv * 16 + hi * 8);
    const float colf = (float)(key0 + l15);
#pragma unroll
    for (int r = 0; r < 8; ++r) {
      float rw = rowf + (float)r;
      float best = -1e30f;
#pragma unroll
      for (int j = 0; j < 4; ++j) {
        float valx = sc[j][r] * 0.125f - slope * fabsf(rw - (colf + 16.0f * j));
        sc[j][r] = valx;
        best = fmaxf(best, valx);
      }
#pragma unroll
      for (int m = 1; m <= 8; m <<= 1) best = fmaxf(best, __shfl_xor(best, m, 32));
      float mn = fmaxf(mrow[r], best);
      float alpha = __expf(mrow[r] - mn);
      mrow[r] = mn;
      lrow[r] = lrow[r] * alpha;
#pragma unroll
      for (int dj = 0; dj < 4; ++dj) o[dj][r] = o[dj][r] * alpha;
      float psum = 0.0f;
#pragma unroll
      for (int j = 0; j < 4; ++j) {
        float pe = __expf(sc[j][r] - mn);
        psum += pe;
        Ps[wv][hi * 8 + r][j * 16 + l15] = f2bf(pe);   // C-layout -> LDS
      }
#pragma unroll
      for (int m = 1; m <= 8; m <<= 1) psum += __shfl_xor(psum, m, 32);
      lrow[r] += psum;
    }

    // O += P * V  (contraction over keys; B[k=key][n=d] read from Vs[d][key])
    Frag pa[2];
#pragma unroll
    for (int kc = 0; kc < 2; ++kc) pa[kc].v = a_frag_lds(&Ps[wv][l15][32 * kc], koff);
#pragma unroll
    for (int dj = 0; dj < 4; ++dj) {
#pragma unroll
      for (int kc = 0; kc < 2; ++kc) {
        v16bf vb = b_frag_lds(&Vs[dj * 16 + l15][32 * kc], kb16);
        o[dj] = __builtin_amdgcn_wmma_f32_16x16x32_bf16(false, pa[kc].v, false, vb, (short)0, o[dj], false, false);
      }
    }
    __syncthreads();
  }

  // normalize and write attn output transposed back into [B,N,C] bf16
#pragma unroll
  for (int r = 0; r < 8; ++r) {
    float inv = 1.0f / lrow[r];
    int nrow = q0 + wv * 16 + hi * 8 + r;
    size_t base = ((size_t)bidx * 1024 + nrow) * 768 + (size_t)hidx * 64;
#pragma unroll
    for (int dj = 0; dj < 4; ++dj)
      AOp[base + dj * 16 + l15] = f2bf(o[dj][r] * inv);
  }
}

// ---------------------------------------------------------------------------
extern "C" void kernel_launch(void* const* d_in, const int* in_sizes, int n_in,
                              void* d_out, int out_size, void* d_ws, size_t ws_size,
                              hipStream_t stream) {
  (void)in_sizes; (void)n_in; (void)out_size; (void)ws_size;
  const float* x      = (const float*)d_in[0];
  const float* w_qkv  = (const float*)d_in[1];
  const float* b_qkv  = (const float*)d_in[2];
  const float* w_proj = (const float*)d_in[3];
  const float* b_proj = (const float*)d_in[4];
  float* out = (float*)d_out;

  // workspace: q,k,v bf16 [B,H,N,D] + attn-out bf16 [B,N,C]  (~100 MB)
  const size_t E = (size_t)16 * 12 * 1024 * 64;   // == 16384*768
  unsigned short* qb = (unsigned short*)d_ws;
  unsigned short* kb = qb + E;
  unsigned short* vb = kb + E;
  unsigned short* ao = vb + E;

  dim3 blk(256);
  gemm_wmma_bf16<0><<<dim3(2304 / 64, 16384 / 128), blk, 0, stream>>>(
      x, nullptr, w_qkv, b_qkv, 16384, 2304, 768, qb, kb, vb, nullptr);
  flash_attn_alibi<<<dim3(1024 / 128, 16 * 12), blk, 0, stream>>>(qb, kb, vb, ao);
  gemm_wmma_bf16<1><<<dim3(768 / 64, 16384 / 128), blk, 0, stream>>>(
      nullptr, ao, w_proj, b_proj, 16384, 768, 768, nullptr, nullptr, nullptr, out);
}
